// MultiHeadedAttention_11656541241626
// MI455X (gfx1250) — matrix operand order repaired and checked by reading, not compiled, not run on hardware
//
#include <hip/hip_runtime.h>
#include <hip/hip_bf16.h>

// ---------------------------------------------------------------------------
// Relation-aware multi-head attention for MI455X (gfx1250), wave32 + WMMA.
// BS=2, L=256, D_MODEL=512, H=8, DK=64.
// Memory-bound on streaming `relation` (268 MB) twice -> K4's relation traffic
// issued as b128 loads via the WMMA A-fragment path; K6 uses strided scalar B loads.
// ---------------------------------------------------------------------------

#define BSZ 2
#define SEQ 256
#define DM  512
#define NH  8
#define DKH 64

typedef __attribute__((ext_vector_type(16))) _Float16 v16h;
typedef __attribute__((ext_vector_type(8)))  _Float16 v8h;
typedef __attribute__((ext_vector_type(8)))  float    v8f;

__device__ __forceinline__ int lane_id() { return (int)(threadIdx.x & 31u); }
__device__ __forceinline__ int wave_id() { return (int)(threadIdx.x >> 5u); }

__device__ __forceinline__ v8f wmma_f16(v16h a, v16h b, v8f c) {
  return __builtin_amdgcn_wmma_f32_16x16x32_f16(
      false, a, false, b, (short)0, c, false, false);
}

// --- A fragment: 16x32, src row-major f32. Row m = lane%16; element e maps to
// k = 8*hi + e (e<8) and 16 + 8*hi + (e-8) (e>=8): two contiguous 8-float runs
// -> two pairs of float4 (b128) loads.
__device__ __forceinline__ v16h load_a_f32(const float* __restrict__ src, int lda,
                                           int row0, int k0, int nrows) {
  int l = lane_id();
  int m = l & 15, hi = l >> 4;
  v16h a;
  if (m < nrows) {
    const float* p = src + (size_t)(row0 + m) * lda + k0 + 8 * hi;
    float4 x0 = *(const float4*)(p);
    float4 x1 = *(const float4*)(p + 4);
    float4 x2 = *(const float4*)(p + 16);
    float4 x3 = *(const float4*)(p + 20);
    a[0] = (_Float16)x0.x; a[1] = (_Float16)x0.y; a[2] = (_Float16)x0.z; a[3] = (_Float16)x0.w;
    a[4] = (_Float16)x1.x; a[5] = (_Float16)x1.y; a[6] = (_Float16)x1.z; a[7] = (_Float16)x1.w;
    a[8] = (_Float16)x2.x; a[9] = (_Float16)x2.y; a[10] = (_Float16)x2.z; a[11] = (_Float16)x2.w;
    a[12] = (_Float16)x3.x; a[13] = (_Float16)x3.y; a[14] = (_Float16)x3.z; a[15] = (_Float16)x3.w;
  } else {
#pragma unroll
    for (int e = 0; e < 16; ++e) a[e] = (_Float16)0.0f;
  }
  return a;
}

// --- A fragment from f16 source (LDS staging): two 16-byte loads per lane.
__device__ __forceinline__ v16h load_a_f16(const _Float16* __restrict__ src, int lda,
                                           int row0, int k0, int nrows) {
  int l = lane_id();
  int m = l & 15, hi = l >> 4;
  v16h a;
  if (m < nrows) {
    const _Float16* p = src + (size_t)(row0 + m) * lda + k0 + 8 * hi;
    v8h lo = *(const v8h*)(p);
    v8h hh = *(const v8h*)(p + 16);
#pragma unroll
    for (int e = 0; e < 8; ++e) { a[e] = lo[e]; a[8 + e] = hh[e]; }
  } else {
#pragma unroll
    for (int e = 0; e < 16; ++e) a[e] = (_Float16)0.0f;
  }
  return a;
}

// --- B fragment: 32x16, B[k][n] = src[k*ldb + n]; column n = lane%16, element e
// -> k = 8*hi + e (e<8), 16 + 8*hi + (e-8) (e>=8): strided scalar loads.
__device__ __forceinline__ v16h load_b_k(const float* __restrict__ src, int ldb,
                                         int k0, int n0) {
  int l = lane_id();
  int n = l & 15, hi = l >> 4;
  const float* p = src + (size_t)(k0 + 8 * hi) * ldb + n0 + n;
  v16h b;
#pragma unroll
  for (int e = 0; e < 16; ++e) b[e] = (_Float16)p[(size_t)(e < 8 ? e : e + 8) * ldb];
  return b;
}

// --- B fragment, transposed f32 source (weights, K^T): scalar strided loads.
__device__ __forceinline__ v16h load_b_t(const float* __restrict__ src, int ldb,
                                         int k0, int n0) {
  int l = lane_id();
  int n = l & 15, hi = l >> 4;
  const float* p = src + (size_t)(n0 + n) * ldb + k0 + 8 * hi;
  v16h b;
#pragma unroll
  for (int e = 0; e < 16; ++e) b[e] = (_Float16)p[e < 8 ? e : e + 8];
  return b;
}

// --- B fragment, transposed f16 source (LDS), with only `ncols` valid columns.
__device__ __forceinline__ v16h load_b_t_f16(const _Float16* __restrict__ src, int ldb,
                                             int k0, int ncols) {
  int l = lane_id();
  int n = l & 15, hi = l >> 4;
  const _Float16* p = src + (size_t)n * ldb + k0 + 8 * hi;
  v16h b;
#pragma unroll
  for (int e = 0; e < 16; ++e)
    b[e] = (n < ncols) ? p[e < 8 ? e : e + 8] : (_Float16)0.0f;
  return b;
}

// --- D store: lane n = lane%16, m = (lane/16)*8 + vgpr
__device__ __forceinline__ void store_d(float* __restrict__ dst, int ldd, int row0,
                                        int n0, v8f d, int nrows,
                                        const float* __restrict__ bias) {
  int l = lane_id();
  int n = l & 15, mh = l >> 4;
  float bv = bias ? bias[n0 + n] : 0.0f;
#pragma unroll
  for (int v = 0; v < 8; ++v) {
    int m = mh * 8 + v;
    if (m < nrows) dst[(size_t)(row0 + m) * ldd + n0 + n] = d[v] + bv;
  }
}

__device__ __forceinline__ void store_d_add(float* __restrict__ dst, int ldd, int row0,
                                            int n0, v8f d, int nrows,
                                            const float* __restrict__ bias) {
  int l = lane_id();
  int n = l & 15, mh = l >> 4;
  float bv = bias ? bias[n0 + n] : 0.0f;
#pragma unroll
  for (int v = 0; v < 8; ++v) {
    int m = mh * 8 + v;
    if (m < nrows) dst[(size_t)(row0 + m) * ldd + n0 + n] += d[v] + bv;
  }
}

// ---------------------------------------------------------------------------
// K1/K8: Y(512x512) = X(512x512) @ W^T + bias.  grid(16,8), block 256 (8 waves).
// ---------------------------------------------------------------------------
__global__ __launch_bounds__(256) void proj_kernel(const float* __restrict__ X,
                                                   const float* __restrict__ W,
                                                   const float* __restrict__ bias,
                                                   float* __restrict__ Y) {
  int w = wave_id();
  int r0 = blockIdx.x * 32 + (w & 1) * 16;
  int c0 = blockIdx.y * 64 + (w >> 1) * 16;
  v8f acc = {0.f, 0.f, 0.f, 0.f, 0.f, 0.f, 0.f, 0.f};
  for (int k0 = 0; k0 < DM; k0 += 32) {
    v16h a = load_a_f32(X, DM, r0, k0, 16);
    v16h b = load_b_t(W, DM, k0, c0);
    acc = wmma_f16(a, b, acc);
  }
  store_d(Y, DM, r0, c0, acc, 16, bias);
}

// ---------------------------------------------------------------------------
// K2: U[h][t][c] = sum_d Q[t][h*64+d] * Wk[h*64+d][c].  grid(16,8,NH).
// ---------------------------------------------------------------------------
__global__ __launch_bounds__(256) void u_kernel(const float* __restrict__ Q,
                                                const float* __restrict__ Wk,
                                                float* __restrict__ U) {
  int h = blockIdx.z;
  int w = wave_id();
  int r0 = blockIdx.x * 32 + (w & 1) * 16;
  int c0 = blockIdx.y * 64 + (w >> 1) * 16;
  v8f acc = {0.f, 0.f, 0.f, 0.f, 0.f, 0.f, 0.f, 0.f};
  for (int k0 = 0; k0 < DKH; k0 += 32) {
    v16h a = load_a_f32(Q, DM, r0, h * DKH + k0, 16);
    v16h b = load_b_k(Wk, DM, h * DKH + k0, c0);
    acc = wmma_f16(a, b, acc);
  }
  store_d(U + (size_t)h * (BSZ * SEQ) * DM, DM, r0, c0, acc, 16, nullptr);
}

// ---------------------------------------------------------------------------
// K3: SC[(b,h)][i][j] = q_i^h . k_j^h.  grid(8,4,BSZ*NH).
// ---------------------------------------------------------------------------
__global__ __launch_bounds__(256) void qk_kernel(const float* __restrict__ Q,
                                                 const float* __restrict__ Kb,
                                                 float* __restrict__ SC) {
  int z = blockIdx.z;
  int b = z >> 3, h = z & 7;
  int w = wave_id();
  int r0 = blockIdx.x * 32 + (w & 1) * 16;
  int c0 = blockIdx.y * 64 + (w >> 1) * 16;
  const float* qb = Q + (size_t)b * SEQ * DM;
  const float* kb = Kb + (size_t)b * SEQ * DM;
  v8f acc = {0.f, 0.f, 0.f, 0.f, 0.f, 0.f, 0.f, 0.f};
  for (int k0 = 0; k0 < DKH; k0 += 32) {
    v16h a = load_a_f32(qb, DM, r0, h * DKH + k0, 16);
    v16h bf = load_b_t(kb, DM, h * DKH + k0, c0);
    acc = wmma_f16(a, bf, acc);
  }
  store_d(SC + (size_t)z * SEQ * SEQ, SEQ, r0, c0, acc, 16, nullptr);
}

// ---------------------------------------------------------------------------
// K4 (heavy, relation stream #1): SC[b,h,i,j] += U[b,h,i,:] . rel[b,i,j,:]
// Reoriented so relation is the A matrix (rows = j, contiguous K = c):
//   D[m=j][n=h] = sum_c rel[j][c] * U[h][c];  U staged in LDS as f16.
// One block per (i,b); 8 waves x 2 j-tiles = 16 M-tiles, K=512, N=16 (8 valid).
// ---------------------------------------------------------------------------
__global__ __launch_bounds__(256) void relscore_kernel(const float* __restrict__ U,
                                                       const float* __restrict__ rel,
                                                       float* __restrict__ SC) {
  int i = blockIdx.x, b = blockIdx.y;
  __shared__ _Float16 sU[NH * DM];  // 8 KB
  for (int idx = threadIdx.x; idx < NH * DM; idx += 256) {
    int h = idx >> 9, c = idx & 511;
    sU[idx] = (_Float16)U[((size_t)h * (BSZ * SEQ) + b * SEQ + i) * DM + c];
  }
  __syncthreads();
  int w = wave_id();
  int l = lane_id();
  const float* rb = rel + (size_t)(b * SEQ + i) * SEQ * DM;
  int j0 = w * 16, j1 = 128 + w * 16;
  v8f acc0 = {0.f, 0.f, 0.f, 0.f, 0.f, 0.f, 0.f, 0.f};
  v8f acc1 = acc0;
#pragma unroll 2
  for (int k0 = 0; k0 < DM; k0 += 32) {
    if (k0 + 64 < DM) {  // stream-ahead prefetch of the relation rows
      __builtin_prefetch(rb + (size_t)(j0 + (l & 15)) * DM + k0 + 64, 0, 1);
      __builtin_prefetch(rb + (size_t)(j1 + (l & 15)) * DM + k0 + 64, 0, 1);
    }
    v16h bf = load_b_t_f16(sU, DM, k0, NH);        // B[c][h] = U[h][c]
    v16h a0 = load_a_f32(rb, DM, j0, k0, 16);      // contiguous b128 stream
    v16h a1 = load_a_f32(rb, DM, j1, k0, 16);
    acc0 = wmma_f16(a0, bf, acc0);
    acc1 = wmma_f16(a1, bf, acc1);
  }
  // D[m=j][n=h]: lane n = h (valid < 8), rows j0 + mh*8 + v (8 consecutive j).
  int h = l & 15, mh = l >> 4;
  if (h < NH) {
    float* base = SC + (size_t)(b * NH + h) * SEQ * SEQ + i * SEQ;
#pragma unroll
    for (int v = 0; v < 8; ++v) base[j0 + mh * 8 + v] += acc0[v];
#pragma unroll
    for (int v = 0; v < 8; ++v) base[j1 + mh * 8 + v] += acc1[v];
  }
}

// ---------------------------------------------------------------------------
// K5: masked, scaled softmax over j.  grid = BSZ*NH*SEQ rows, block 256.
// ---------------------------------------------------------------------------
__global__ __launch_bounds__(256) void softmax_kernel(float* __restrict__ SC,
                                                      const int* __restrict__ mask) {
  int r = blockIdx.x;  // (b*8+h)*256 + i
  int b = r >> 11;
  int i = r & 255;
  int j = threadIdx.x;
  __shared__ float red[256];
  float v = SC[(size_t)r * SEQ + j] * 0.125f;  // 1/sqrt(64)
  if (mask[(size_t)b * SEQ * SEQ + i * SEQ + j] == 0) v = -1e9f;
  red[j] = v;
  __syncthreads();
  for (int s = 128; s > 0; s >>= 1) {
    if (j < s) red[j] = fmaxf(red[j], red[j + s]);
    __syncthreads();
  }
  float mx = red[0];
  __syncthreads();
  float e = __expf(v - mx);
  red[j] = e;
  __syncthreads();
  for (int s = 128; s > 0; s >>= 1) {
    if (j < s) red[j] += red[j + s];
    __syncthreads();
  }
  SC[(size_t)r * SEQ + j] = e / red[0];
}

// ---------------------------------------------------------------------------
// K6 (heavy, relation stream #2): S[b,h,i,c] = sum_j attn[b,h,i,j]*rel[b,i,j,c]
// attn staged in LDS as f16; relation streamed row-major through load_b_k
// (16 strided scalar loads per fragment). One block per (i,b); M=8 pad, K=256, N=512.
// ---------------------------------------------------------------------------
__global__ __launch_bounds__(256) void srel_kernel(const float* __restrict__ SC,
                                                   const float* __restrict__ rel,
                                                   float* __restrict__ S) {
  int i = blockIdx.x, b = blockIdx.y;
  __shared__ _Float16 sA[NH * SEQ];  // 4 KB
  for (int idx = threadIdx.x; idx < NH * SEQ; idx += 256) {
    int h = idx >> 8, j = idx & 255;
    sA[idx] = (_Float16)SC[((size_t)(b * NH + h) * SEQ + i) * SEQ + j];
  }
  __syncthreads();
  int w = wave_id();
  int l = lane_id();
  const float* rb = rel + (size_t)(b * SEQ + i) * SEQ * DM;
  v8f acc[4];
#pragma unroll
  for (int p = 0; p < 4; ++p) acc[p] = v8f{0.f, 0.f, 0.f, 0.f, 0.f, 0.f, 0.f, 0.f};
#pragma unroll 2
  for (int k0 = 0; k0 < SEQ; k0 += 32) {
    if (k0 + 32 < SEQ)
      __builtin_prefetch(rb + (size_t)(k0 + 32 + l) * DM + w * 16, 0, 1);
    v16h a = load_a_f16(sA, SEQ, 0, k0, NH);
#pragma unroll
    for (int p = 0; p < 4; ++p) {
      v16h bf = load_b_k(rb, DM, k0, w * 16 + p * 128);
      acc[p] = wmma_f16(a, bf, acc[p]);
    }
  }
  float* sp = S + (size_t)b * NH * SEQ * DM + i * DM;  // + h*SEQ*DM per row
  for (int p = 0; p < 4; ++p)
    store_d(sp, SEQ * DM, 0, w * 16 + p * 128, acc[p], NH, nullptr);
}

// ---------------------------------------------------------------------------
// K7a: X[b,i,h*64+d] = sum_j attn[b,h,i,j] * V[b,j,h*64+d].  grid(8, BSZ*NH).
// ---------------------------------------------------------------------------
__global__ __launch_bounds__(256) void av_kernel(const float* __restrict__ SC,
                                                 const float* __restrict__ V,
                                                 float* __restrict__ X) {
  int z = blockIdx.y;  // b*8+h
  int b = z >> 3, h = z & 7;
  int w = wave_id();
  int r0 = blockIdx.x * 32 + (w & 1) * 16;
  int c0 = (w >> 1) * 16;
  const float* a_src = SC + (size_t)z * SEQ * SEQ;
  const float* b_src = V + (size_t)b * SEQ * DM + h * DKH;
  v8f acc = {0.f, 0.f, 0.f, 0.f, 0.f, 0.f, 0.f, 0.f};
  for (int k0 = 0; k0 < SEQ; k0 += 32) {
    v16h a = load_a_f32(a_src, SEQ, r0, k0, 16);
    v16h bf = load_b_k(b_src, DM, k0, c0);
    acc = wmma_f16(a, bf, acc);
  }
  store_d(X + (size_t)b * SEQ * DM + h * DKH, DM, r0, c0, acc, 16, nullptr);
}

// ---------------------------------------------------------------------------
// K7b: X[t, h*64+d] += sum_c S[b,h,i,c]*Wv[h*64+d,c]  (+ bv, the rv bias).
// ---------------------------------------------------------------------------
__global__ __launch_bounds__(256) void svw_kernel(const float* __restrict__ S,
                                                  const float* __restrict__ Wv,
                                                  const float* __restrict__ bv,
                                                  float* __restrict__ X) {
  int h = blockIdx.y;
  int w = wave_id();
  int r0 = blockIdx.x * 32 + (w & 1) * 16;
  int c0 = (w >> 1) * 16;
  int b = r0 >> 8, i0 = r0 & 255;
  const float* a_src = S + (size_t)(b * NH + h) * SEQ * DM;
  v8f acc = {0.f, 0.f, 0.f, 0.f, 0.f, 0.f, 0.f, 0.f};
  for (int k0 = 0; k0 < DM; k0 += 32) {
    v16h a = load_a_f32(a_src, DM, i0, k0, 16);
    v16h bf = load_b_t(Wv + (size_t)h * DKH * DM, DM, k0, c0);
    acc = wmma_f16(a, bf, acc);
  }
  store_d_add(X + h * DKH, DM, r0, c0, acc, 16, bv + h * DKH);
}

// ---------------------------------------------------------------------------
// Launch
// ---------------------------------------------------------------------------
extern "C" void kernel_launch(void* const* d_in, const int* in_sizes, int n_in,
                              void* d_out, int out_size, void* d_ws, size_t ws_size,
                              hipStream_t stream) {
  (void)in_sizes; (void)n_in; (void)out_size; (void)ws_size;
  const float* query = (const float*)d_in[0];
  const float* key   = (const float*)d_in[1];
  const float* value = (const float*)d_in[2];
  const float* rel   = (const float*)d_in[3];
  const int*   mask  = (const int*)d_in[4];
  const float* Wq = (const float*)d_in[5];  const float* bq = (const float*)d_in[6];
  const float* Wk = (const float*)d_in[7];  const float* bk = (const float*)d_in[8];
  const float* Wv = (const float*)d_in[9];  const float* bv = (const float*)d_in[10];
  const float* Wo = (const float*)d_in[11]; const float* bo = (const float*)d_in[12];
  float* out = (float*)d_out;

  float* w = (float*)d_ws;
  float* Q  = w;
  float* Kb = Q  + (size_t)BSZ * SEQ * DM;
  float* Vb = Kb + (size_t)BSZ * SEQ * DM;
  float* U  = Vb + (size_t)BSZ * SEQ * DM;        // NH*BSZ*SEQ*DM
  float* SC = U  + (size_t)NH * BSZ * SEQ * DM;   // BSZ*NH*SEQ*SEQ
  float* S  = SC + (size_t)BSZ * NH * SEQ * SEQ;  // BSZ*NH*SEQ*DM
  float* X  = S  + (size_t)BSZ * NH * SEQ * DM;   // BSZ*SEQ*DM

  dim3 blk(256);
  proj_kernel<<<dim3(16, 8), blk, 0, stream>>>(query, Wq, bq, Q);
  proj_kernel<<<dim3(16, 8), blk, 0, stream>>>(key,   Wk, bk, Kb);
  proj_kernel<<<dim3(16, 8), blk, 0, stream>>>(value, Wv, bv, Vb);
  u_kernel<<<dim3(16, 8, NH), blk, 0, stream>>>(Q, Wk, U);
  qk_kernel<<<dim3(8, 4, BSZ * NH), blk, 0, stream>>>(Q, Kb, SC);
  relscore_kernel<<<dim3(SEQ, BSZ), blk, 0, stream>>>(U, rel, SC);
  softmax_kernel<<<dim3(BSZ * NH * SEQ), blk, 0, stream>>>(SC, mask);
  srel_kernel<<<dim3(SEQ, BSZ), blk, 0, stream>>>(SC, rel, S);
  av_kernel<<<dim3(8, BSZ * NH), blk, 0, stream>>>(SC, Vb, X);
  svw_kernel<<<dim3(16, NH), blk, 0, stream>>>(S, Wv, bv, X);
  proj_kernel<<<dim3(16, 8), blk, 0, stream>>>(X, Wo, bo, out);
}
